// SHDSNN_15831249453639
// MI455X (gfx1250) — matrix operand order repaired and checked by reading, not compiled, not run on hardware
//
#include <hip/hip_runtime.h>
#include <math.h>

typedef __attribute__((ext_vector_type(16))) __bf16 v16bf;
typedef __attribute__((ext_vector_type(8)))  float  v8f;
typedef unsigned int u32;

#define T_STEPS 250
#define BATCH   256
#define NIN     700
#define NINP    704            // K padded to multiple of 32
#define NHID    1024
#define NOUT    20
#define ROWS    (T_STEPS * BATCH)   // 64000
#define P2_BLOCKS 64

struct __align__(16) U4 { u32 x, y, z, w; };
union  FragU { U4 u[2]; v16bf v; };
union  BfU4  { U4 u; __bf16 s[8]; };

__device__ __forceinline__ v16bf frag2(U4 a, U4 b) {
  FragU f; f.u[0] = a; f.u[1] = b; return f.v;
}

#if defined(__HIP_DEVICE_COMPILE__) && __has_builtin(__builtin_amdgcn_global_load_async_to_lds_b128)
#define USE_ASYNC_LDS 1
#else
#define USE_ASYNC_LDS 0
#endif

#if USE_ASYNC_LDS
typedef int vint4 __attribute__((vector_size(16)));
typedef __attribute__((address_space(1))) vint4 gvec4;  // global (AS1)
typedef __attribute__((address_space(3))) vint4 lvec4;  // LDS (AS3)
#endif

__device__ __forceinline__ void async_wait0() {
#if USE_ASYNC_LDS
#if __has_builtin(__builtin_amdgcn_s_wait_asynccnt)
  __builtin_amdgcn_s_wait_asynccnt(0);
#else
  asm volatile("s_wait_asynccnt 0" ::: "memory");
#endif
#endif
}

// allow the prefetched stage (4 loads/thread) to remain in flight
__device__ __forceinline__ void async_wait4() {
#if USE_ASYNC_LDS
#if __has_builtin(__builtin_amdgcn_s_wait_asynccnt)
  __builtin_amdgcn_s_wait_asynccnt(4);
#else
  asm volatile("s_wait_asynccnt 4" ::: "memory");
#endif
#endif
}

__device__ __forceinline__ void copy16(const U4* src, U4* dst) {
#if USE_ASYNC_LDS
  U4* s = const_cast<U4*>(src);
  __builtin_amdgcn_global_load_async_to_lds_b128((gvec4*)s, (lvec4*)dst, 0, 0);
#else
  *dst = *src;
#endif
}

// ---- grid-wide sense-reversing barrier (persistent kernel) ----
__device__ __forceinline__ void grid_barrier(u32* cnt, u32* gen, u32 nblk) {
  __syncthreads();
  if (threadIdx.x == 0) {
    u32 g = __hip_atomic_load(gen, __ATOMIC_RELAXED, __HIP_MEMORY_SCOPE_AGENT);
    u32 t = __hip_atomic_fetch_add(cnt, 1u, __ATOMIC_ACQ_REL, __HIP_MEMORY_SCOPE_AGENT);
    if (t == nblk - 1u) {
      __hip_atomic_store(cnt, 0u, __ATOMIC_RELAXED, __HIP_MEMORY_SCOPE_AGENT);
      __hip_atomic_fetch_add(gen, 1u, __ATOMIC_RELEASE, __HIP_MEMORY_SCOPE_AGENT);
    } else {
      while (__hip_atomic_load(gen, __ATOMIC_ACQUIRE, __HIP_MEMORY_SCOPE_AGENT) == g)
        __builtin_amdgcn_s_sleep(2);
    }
  }
  __syncthreads();
}

// ================= kernel 0a: convert weights to bf16, zero out/barrier ======
__global__ void k_conv_w(const float* __restrict__ W1, const float* __restrict__ Wrec,
                         const float* __restrict__ Wout,
                         __bf16* __restrict__ W1b, __bf16* __restrict__ Wrecb,
                         __bf16* __restrict__ Woutb, float* __restrict__ out,
                         u32* __restrict__ bar) {
  int i0 = blockIdx.x * blockDim.x + threadIdx.x;
  int stride = gridDim.x * blockDim.x;
  for (int i = i0; i < NHID * NINP; i += stride) {
    int h = i / NINP, k = i - h * NINP;
    W1b[i] = (__bf16)((k < NIN) ? W1[h * NIN + k] : 0.0f);
  }
  for (int i = i0; i < NHID * NHID; i += stride) Wrecb[i] = (__bf16)Wrec[i];
  for (int i = i0; i < NOUT * NHID; i += stride) Woutb[i] = (__bf16)Wout[i];
  for (int i = i0; i < BATCH * NOUT; i += stride) out[i] = 0.0f;
  if (i0 < 2) bar[i0] = 0u;
}

// ================= kernel 0b: convert+permute x -> bf16 [t*B+b][NINP] ========
__global__ void k_conv_x(const float* __restrict__ x, __bf16* __restrict__ xb) {
  int i0 = blockIdx.x * blockDim.x + threadIdx.x;
  int stride = gridDim.x * blockDim.x;
  const int total = ROWS * NINP;
  for (int i = i0; i < total; i += stride) {
    int r = i / NINP, k = i - r * NINP;
    int t = r >> 8, b = r & 255;
    float v = (k < NIN) ? x[((size_t)b * T_STEPS + t) * NIN + k] : 0.0f;
    xb[i] = (__bf16)v;
  }
}

// ================= kernel 1: Ix = Xbf @ W1bf^T  (bf16 WMMA, fp32 acc) ========
// C tile 128x128 per block, 8 waves, wave tile 32x64, K-step 32.
// Double-buffered async-to-LDS staging: stage k+1 DMA overlaps stage k WMMAs.
__device__ __forceinline__ void stage_tiles(const __bf16* __restrict__ A,
                                            const __bf16* __restrict__ Bw,
                                            __bf16* As, __bf16* Bs,
                                            int rowBase, int colBase, int k0, int tid) {
#pragma unroll
  for (int q = 0; q < 2; q++) {          // 512 16B chunks each, 2 per thread
    int c = tid + q * 256;
    int r = c >> 2, o = c & 3;
    copy16((const U4*)(A  + (size_t)(rowBase + r) * NINP + k0) + o, (U4*)(As + r * 32) + o);
    copy16((const U4*)(Bw + (size_t)(colBase + r) * NINP + k0) + o, (U4*)(Bs + r * 32) + o);
  }
}

__global__ void __launch_bounds__(256)
k_in_gemm(const __bf16* __restrict__ A,   // [ROWS][NINP]
          const __bf16* __restrict__ Bw,  // [NHID][NINP]
          __bf16* __restrict__ Cc) {      // [ROWS][NHID] bf16
  extern __shared__ char smem[];
  __bf16* AsB[2] = { (__bf16*)smem,           (__bf16*)(smem + 16384) };
  __bf16* BsB[2] = { (__bf16*)(smem + 8192),  (__bf16*)(smem + 24576) };
  const int tid  = threadIdx.x;
  const int w    = tid >> 5, lane = tid & 31;
  const int mg   = w & 3, ng = w >> 2;        // wave: rows mg*32.., cols ng*64..
  const int rowBase = blockIdx.x * 128;
  const int colBase = blockIdx.y * 128;
  const int lhi = lane >> 4, llo = lane & 15;
  const int NK = NINP / 32;                   // 22

  v8f zero = {};
  v8f acc[2][4];
#pragma unroll
  for (int i = 0; i < 2; i++)
#pragma unroll
    for (int j = 0; j < 4; j++) acc[i][j] = zero;

  stage_tiles(A, Bw, AsB[0], BsB[0], rowBase, colBase, 0, tid);
  for (int it = 0; it < NK; it++) {
    const int cur = it & 1;
    if (it + 1 < NK) {
      stage_tiles(A, Bw, AsB[cur ^ 1], BsB[cur ^ 1], rowBase, colBase, (it + 1) * 32, tid);
      async_wait4();                          // stage `it` landed; prefetch in flight
    } else {
      async_wait0();
    }
    __syncthreads();
    const __bf16* As = AsB[cur];
    const __bf16* Bs = BsB[cur];
#pragma unroll
    for (int rt = 0; rt < 2; rt++) {
      int ar = mg * 32 + rt * 16 + llo;
      const U4* ap = (const U4*)(As + ar * 32 + lhi * 8);
      v16bf afr = frag2(ap[0], ap[2]);   // K 0..7 / 16..23 (lanes<16), +8 (lanes>=16)
#pragma unroll
      for (int ct = 0; ct < 4; ct++) {
        int brow = ng * 64 + ct * 16 + llo;
        const U4* bp = (const U4*)(Bs + brow * 32 + lhi * 8);
        v16bf bfr = frag2(bp[0], bp[2]); // K 0..7 / 16..23 (lanes<16), +8 (lanes>=16)
        acc[rt][ct] = __builtin_amdgcn_wmma_f32_16x16x32_bf16(
            false, afr, false, bfr, (short)0, acc[rt][ct], false, false);
      }
    }
    __syncthreads();                          // done reading `cur` before reuse
  }
#pragma unroll
  for (int rt = 0; rt < 2; rt++)
#pragma unroll
    for (int ct = 0; ct < 4; ct++)
#pragma unroll
      for (int i = 0; i < 8; i++) {
        int r = rowBase + mg * 32 + rt * 16 + lhi * 8 + i;
        int h = colBase + ng * 64 + ct * 16 + llo;
        Cc[(size_t)r * NHID + h] = (__bf16)acc[rt][ct][i];
      }
}

// ================= kernel 2: persistent recurrent adLIF loop =================
// 64 blocks x 256 threads; block owns fixed 64x64 (b,h) tile; Wrec slice in LDS;
// v1/a1 state in registers; spikes through global (L2) with grid barrier per step.
__global__ void __launch_bounds__(256)
k_recurrent(const __bf16* __restrict__ Wrecb,  // [NHID][NHID]
            const __bf16* __restrict__ Ixb,    // [ROWS][NHID], r = t*256+b
            __bf16* __restrict__ spk,          // [(T+1)][BATCH][NHID]
            const float* __restrict__ alpha, const float* __restrict__ rho,
            const float* __restrict__ beta_a, u32* __restrict__ bar) {
  extern __shared__ char smem[];
  __bf16* Ws = (__bf16*)smem;                  // [64][1024] Wrec col-slice, 128KB
  const int tid = threadIdx.x;
  const int w = tid >> 5, lane = tid & 31;
  const int lhi = lane >> 4, llo = lane & 15;
  const int blk = blockIdx.x;
  const int rowBase = (blk & 3) * 64;          // batch rows
  const int colBase = (blk >> 2) * 64;         // hidden cols
  const int mg = w & 3, ng = w >> 2;           // wave: rows mg*16, cols ng*32

  // load Wrec slice [colBase..+64)[0..1024) into LDS: 8192 16B chunks
  for (int c = tid; c < 8192; c += 256) {
    int r = c >> 7, o = c & 127;
    ((U4*)(Ws + r * 1024))[o] = ((const U4*)(Wrecb + (size_t)(colBase + r) * 1024))[o];
  }
  // zero spk slot 0 cooperatively (this region aliases consumed xb scratch)
  U4 z4; z4.x = z4.y = z4.z = z4.w = 0u;
  for (int c = blk * 256 + tid; c < (BATCH * NHID * 2) / 16; c += P2_BLOCKS * 256)
    ((U4*)spk)[c] = z4;
  grid_barrier(bar, bar + 1, P2_BLOCKS);

  // per-lane neuron parameters (constant over time)
  int   hh[2]; float al[2], oma[2], rh[2], ba[2];
#pragma unroll
  for (int ct = 0; ct < 2; ct++) {
    int h = colBase + ng * 32 + ct * 16 + llo;
    hh[ct] = h; al[ct] = alpha[h]; oma[ct] = 1.0f - al[ct];
    rh[ct] = rho[h]; ba[ct] = beta_a[h];
  }
  float v1[2][8], a1[2][8];
#pragma unroll
  for (int ct = 0; ct < 2; ct++)
#pragma unroll
    for (int i = 0; i < 8; i++) { v1[ct][i] = 0.0f; a1[ct][i] = 0.0f; }

  const int ar = rowBase + mg * 16 + llo;      // A (spike) row for this lane
  const int bb = rowBase + mg * 16 + lhi * 8;  // first batch row of this lane's C slots
  v8f zero = {};

  for (int t = 0; t < T_STEPS; t++) {
    const __bf16* sp  = spk + (size_t)t * BATCH * NHID;
    const __bf16* ixr = Ixb + (size_t)t * BATCH * NHID;
    // hide Ix latency behind the K-loop (global_prefetch_b8)
#pragma unroll
    for (int ct = 0; ct < 2; ct++)
      __builtin_prefetch(&ixr[(size_t)bb * NHID + hh[ct]], 0, 0);

    v8f acc[2]; acc[0] = zero; acc[1] = zero;
#pragma unroll 4
    for (int k0 = 0; k0 < NHID; k0 += 32) {
      const U4* ap = (const U4*)(sp + (size_t)ar * NHID + k0 + lhi * 8);
      v16bf afr = frag2(ap[0], ap[2]);
#pragma unroll
      for (int ct = 0; ct < 2; ct++) {
        int brow = ng * 32 + ct * 16 + llo;
        const U4* bp = (const U4*)(Ws + brow * 1024 + k0 + lhi * 8);
        v16bf bfr = frag2(bp[0], bp[2]);
        acc[ct] = __builtin_amdgcn_wmma_f32_16x16x32_bf16(
            false, afr, false, bfr, (short)0, acc[ct], false, false);
      }
    }
    // adLIF update (state in registers), emit spikes for step t+1
    __bf16* spn = spk + (size_t)(t + 1) * BATCH * NHID;
#pragma unroll
    for (int ct = 0; ct < 2; ct++) {
      int h = hh[ct];
#pragma unroll
      for (int i = 0; i < 8; i++) {
        int b = bb + i;
        float I1 = acc[ct][i] + (float)ixr[(size_t)b * NHID + h];
        float sprev = v1[ct][i] > 1.0f ? 1.0f : 0.0f;   // spike from previous step
        float vn = al[ct] * (v1[ct][i] - sprev) + oma[ct] * (I1 - a1[ct][i]);
        float s  = vn > 1.0f ? 1.0f : 0.0f;
        a1[ct][i] = rh[ct] * a1[ct][i] + ba[ct] * s;
        v1[ct][i] = vn;
        spn[(size_t)b * NHID + h] = (__bf16)s;
      }
    }
    grid_barrier(bar, bar + 1, P2_BLOCKS);
  }
}

// ================= kernel 3: readout folded over time ========================
// out[b,o] = (1/T) * sum_t (1 - beta_o^(T-t)) * (spk_t @ Wout^T)[b,o]
__global__ void __launch_bounds__(256)
k_readout(const __bf16* __restrict__ spk, const __bf16* __restrict__ Woutb,
          const float* __restrict__ beta_out, float* __restrict__ out) {
  __shared__ __bf16 Wl[NOUT * NHID];   // 40KB
  __shared__ float  cs[NOUT];
  const int t = blockIdx.x;
  const int tid = threadIdx.x;
  for (int c = tid; c < (NOUT * NHID) / 8; c += 256)
    ((U4*)Wl)[c] = ((const U4*)Woutb)[c];
  if (tid < NOUT) {
    float b = beta_out[tid];
    cs[tid] = (1.0f - powf(b, (float)(T_STEPS - t))) / (float)T_STEPS;
  }
  __syncthreads();
  const int b = tid;  // batch row
  const __bf16* srow = spk + ((size_t)(t + 1) * BATCH + b) * NHID;
  float acc[NOUT];
#pragma unroll
  for (int o = 0; o < NOUT; o++) acc[o] = 0.0f;
  for (int h0 = 0; h0 < NHID; h0 += 8) {
    BfU4 v; v.u = *((const U4*)(srow + h0));
#pragma unroll
    for (int j = 0; j < 8; j++) {
      float s = (float)v.s[j];
      if (s != 0.0f) {
        int h = h0 + j;
#pragma unroll
        for (int o = 0; o < NOUT; o++) acc[o] += (float)Wl[o * NHID + h];
      }
    }
  }
#pragma unroll
  for (int o = 0; o < NOUT; o++)
    atomicAdd(&out[b * NOUT + o], cs[o] * acc[o]);
}

// ============================== host launcher ================================
extern "C" void kernel_launch(void* const* d_in, const int* in_sizes, int n_in,
                              void* d_out, int out_size, void* d_ws, size_t ws_size,
                              hipStream_t stream) {
  (void)in_sizes; (void)n_in; (void)out_size; (void)ws_size;
  const float* x        = (const float*)d_in[0];
  const float* W1       = (const float*)d_in[1];
  const float* Wrec     = (const float*)d_in[2];
  const float* Wout     = (const float*)d_in[3];
  const float* alpha    = (const float*)d_in[4];
  const float* rho      = (const float*)d_in[5];
  const float* beta_a   = (const float*)d_in[6];
  const float* beta_out = (const float*)d_in[7];
  float* out = (float*)d_out;

  char* ws = (char*)d_ws;
  size_t off = 0;
  auto carve = [&](size_t bytes) -> void* {
    void* p = (void*)(ws + off);
    off += (bytes + 255) & ~(size_t)255;
    return p;
  };
  __bf16* W1b   = (__bf16*)carve((size_t)NHID * NINP * 2);
  __bf16* Wrecb = (__bf16*)carve((size_t)NHID * NHID * 2);
  __bf16* Woutb = (__bf16*)carve((size_t)NOUT * NHID * 2);
  u32*    bar   = (u32*)carve(256);
  __bf16* Ixb   = (__bf16*)carve((size_t)ROWS * NHID * 2);
  __bf16* spk   = (__bf16*)carve((size_t)(T_STEPS + 1) * BATCH * NHID * 2);
  __bf16* xb    = spk;  // alias: xb fully consumed by GEMM before spk produced

  k_conv_w<<<256, 256, 0, stream>>>(W1, Wrec, Wout, W1b, Wrecb, Woutb, out, bar);
  k_conv_x<<<4096, 256, 0, stream>>>(x, xb);
  k_in_gemm<<<dim3(ROWS / 128, NHID / 128), 256, 4 * 128 * 32 * 2, stream>>>(xb, W1b, Ixb);
  k_recurrent<<<P2_BLOCKS, 256, 64 * 1024 * 2, stream>>>(Wrecb, Ixb, spk,
                                                         alpha, rho, beta_a, bar);
  k_readout<<<T_STEPS, 256, 0, stream>>>(spk, Woutb, beta_out, out);
}
